// MoRELayerSynthesisIOptionA_52888227283726
// MI455X (gfx1250) — matrix operand order repaired and checked by reading, not compiled, not run on hardware
//
#include <hip/hip_runtime.h>

// ---------------------------------------------------------------------------
// MoRE layer for MI455X (gfx1250): bf16 WMMA GEMMs, double-buffered LDS fed by
// CDNA5 async global->LDS DMA (ASYNCcnt), fp32 elementwise fusion.
// ---------------------------------------------------------------------------

typedef __attribute__((ext_vector_type(16))) __bf16 v16bf;
typedef __attribute__((ext_vector_type(8)))  __bf16 v8bf;
typedef __attribute__((ext_vector_type(8)))  float  v8f;

#define T_TOK 4096
#define DMODEL 1024
#define FDIM   2048
#define NEXP   8
#define LDT    40            // padded LDS row stride in bf16 (80B rows, 16B chunks)
#define TILE_B (128 * LDT * 2)  // bytes per LDS tile buffer (10240)

#define ASYNC_TILE_LOADS 1

// 32-bit LDS byte address for inline asm (generic -> addrspace(3) cast).
__device__ __forceinline__ unsigned lds_addr32(const void* p) {
    return (unsigned)(unsigned long long)
        (__attribute__((address_space(3))) const void*)p;
}

#define ASYNC_LOAD_B128(ldsoff, gaddr)                                   \
    asm volatile("global_load_async_to_lds_b128 %0, %1, off"             \
                 :: "v"(ldsoff), "v"(gaddr) : "memory")

// ===========================================================================
// Tiled NT GEMM: C[M,N] = A[M,K] * W[N,K]^T, bf16 inputs, f32 accumulate.
// Block tile 128x128x32, 256 threads = 8 waves (4 along M, 2 along N),
// wave tile 32x64 = 2x4 v_wmma_f32_16x16x32_bf16 accumulators.
// Double-buffered LDS, tiles streamed by async global->LDS (4 ops/thread/tile).
// EPI: 0 -> f32 store, 1 -> bf16 store, 2 -> silu then bf16 store.
// ===========================================================================
template<int EPI>
__global__ __launch_bounds__(256)
void gemm_nt_wmma_kernel(const __bf16* __restrict__ A,
                         const __bf16* __restrict__ W,
                         float* __restrict__ Cf,
                         __bf16* __restrict__ Cb,
                         int M, int N, int K)
{
    __shared__ __bf16 As[2][128 * LDT];
    __shared__ __bf16 Bs[2][128 * LDT];

    const int tid   = threadIdx.x;
    const int lane  = tid & 31;
    const int wave  = tid >> 5;
    const int waveM = wave & 3;     // 0..3
    const int waveN = wave >> 2;    // 0..1
    const int half  = lane >> 4;
    const int l16   = lane & 15;

    const int m0 = blockIdx.y * 128;
    const int n0 = blockIdx.x * 128;

    // Per-thread cooperative-load assignment: 2 x 16B chunks of A, 2 of W.
    const __bf16* gA[2];
    const __bf16* gB[2];
    unsigned ldsA0[2], ldsB0[2];
#pragma unroll
    for (int i = 0; i < 2; ++i) {
        int chunk = tid + i * 256;
        int row   = chunk >> 2;
        int kc    = (chunk & 3) << 3;
        gA[i] = A + (long)(m0 + row) * K + kc;
        gB[i] = W + (long)(n0 + row) * K + kc;
        ldsA0[i] = lds_addr32(&As[0][row * LDT + kc]);
        ldsB0[i] = lds_addr32(&Bs[0][row * LDT + kc]);
    }

    v8f acc[2][4];
#pragma unroll
    for (int i = 0; i < 2; ++i)
#pragma unroll
        for (int j = 0; j < 4; ++j) {
            v8f z = {0.f, 0.f, 0.f, 0.f, 0.f, 0.f, 0.f, 0.f};
            acc[i][j] = z;
        }

    const int nk = K >> 5;

#if ASYNC_TILE_LOADS
    // Prologue: stream tile 0 into buffer 0 (ASYNCcnt += 4 for this wave).
#pragma unroll
    for (int i = 0; i < 2; ++i) {
        ASYNC_LOAD_B128(ldsA0[i], (unsigned long long)gA[i]);
        ASYNC_LOAD_B128(ldsB0[i], (unsigned long long)gB[i]);
    }
#endif

    for (int kt = 0; kt < nk; ++kt) {
        const int b = kt & 1;

#if ASYNC_TILE_LOADS
        if (kt + 1 < nk) {
            const unsigned nboff = (unsigned)((b ^ 1) * TILE_B);
#pragma unroll
            for (int i = 0; i < 2; ++i) {
                gA[i] += 32; gB[i] += 32;
                ASYNC_LOAD_B128(ldsA0[i] + nboff, (unsigned long long)gA[i]);
                ASYNC_LOAD_B128(ldsB0[i] + nboff, (unsigned long long)gB[i]);
            }
            if (kt + 2 < nk) {   // pull tile kt+2 toward L2
                __builtin_prefetch(gA[0] + 32, 0, 3);
                __builtin_prefetch(gB[0] + 32, 0, 3);
            }
            // async loads complete in order: <=4 outstanding => tile kt landed
            asm volatile("s_wait_asynccnt 4" ::: "memory");
        } else {
            asm volatile("s_wait_asynccnt 0" ::: "memory");
        }
#else
        {   // synchronous fallback: stage in registers, then LDS stores
            uint4 ra[2], rb[2];
#pragma unroll
            for (int i = 0; i < 2; ++i) {
                ra[i] = *(const uint4*)gA[i];
                rb[i] = *(const uint4*)gB[i];
            }
#pragma unroll
            for (int i = 0; i < 2; ++i) {
                int chunk = tid + i * 256;
                int row   = chunk >> 2;
                int kc    = (chunk & 3) << 3;
                *(uint4*)(&As[b][row * LDT + kc]) = ra[i];
                *(uint4*)(&Bs[b][row * LDT + kc]) = rb[i];
                gA[i] += 32; gB[i] += 32;
            }
        }
#endif
        __syncthreads();   // tile kt visible to all waves

        // A fragments: ISA 16-bit A 16x32 layout.
        // lanes 0-15: row M=l16, K 0..7 & 16..23; lanes 16-31: K 8..15 & 24..31.
        v16bf a[2], bfr[4];
        const __bf16* Ab = As[b];
        const __bf16* Bb = Bs[b];
#pragma unroll
        for (int mi = 0; mi < 2; ++mi) {
            int row = waveM * 32 + mi * 16 + l16;
            v8bf lo = *(const v8bf*)(&Ab[row * LDT + half * 8]);
            v8bf hi = *(const v8bf*)(&Ab[row * LDT + half * 8 + 16]);
            a[mi] = __builtin_shufflevector(lo, hi, 0, 1, 2, 3, 4, 5, 6, 7,
                                            8, 9, 10, 11, 12, 13, 14, 15);
        }
        // B fragments: column N=l16; lanes 0-15 K 0..7 & 16..23; lanes 16-31: K 8..15 & 24..31.
#pragma unroll
        for (int ni = 0; ni < 4; ++ni) {
            int col = waveN * 64 + ni * 16 + l16;
            v8bf lo = *(const v8bf*)(&Bb[col * LDT + half * 8]);
            v8bf hi = *(const v8bf*)(&Bb[col * LDT + half * 8 + 16]);
            bfr[ni] = __builtin_shufflevector(lo, hi, 0, 1, 2, 3, 4, 5, 6, 7,
                                              8, 9, 10, 11, 12, 13, 14, 15);
        }
#pragma unroll
        for (int mi = 0; mi < 2; ++mi)
#pragma unroll
            for (int ni = 0; ni < 4; ++ni)
                acc[mi][ni] = __builtin_amdgcn_wmma_f32_16x16x32_bf16(
                    false, a[mi], false, bfr[ni], (short)0, acc[mi][ni], false, false);

        __syncthreads();   // everyone done reading buf b before it is re-filled
    }

    // Epilogue: C 16x16 f32 layout -> row = half*8 + r, col = l16.
#pragma unroll
    for (int mi = 0; mi < 2; ++mi)
#pragma unroll
        for (int ni = 0; ni < 4; ++ni)
#pragma unroll
            for (int r = 0; r < 8; ++r) {
                int row = m0 + waveM * 32 + mi * 16 + half * 8 + r;
                int col = n0 + waveN * 64 + ni * 16 + l16;
                float v = acc[mi][ni][r];
                if (EPI == 2) v = v * (1.f / (1.f + __expf(-v)));   // silu
                if (EPI == 0) Cf[(long)row * N + col] = v;
                else          Cb[(long)row * N + col] = (__bf16)v;
            }
}

// ===========================================================================
// Elementwise / reduction kernels
// ===========================================================================
__global__ void zero_f32_kernel(float* p, long n) {
    long i = (long)blockIdx.x * blockDim.x + threadIdx.x;
    long s = (long)gridDim.x * blockDim.x;
    for (; i < n; i += s) p[i] = 0.f;
}

__global__ void copy_f32_kernel(const float* __restrict__ in, float* __restrict__ out, long n) {
    long i = (long)blockIdx.x * blockDim.x + threadIdx.x;
    long s = (long)gridDim.x * blockDim.x;
    for (; i < n; i += s) out[i] = in[i];
}

__global__ void f32_to_bf16_kernel(const float* __restrict__ in, __bf16* __restrict__ out, long n) {
    long i = (long)blockIdx.x * blockDim.x + threadIdx.x;
    long s = (long)gridDim.x * blockDim.x;
    for (; i < n; i += s) out[i] = (__bf16)in[i];
}

// per-row f32->bf16 with independent strides (fills halves of `combined`)
__global__ __launch_bounds__(256)
void conv_rows_kernel(const float* __restrict__ in, int inStride,
                      __bf16* __restrict__ out, int outStride) {
    int t = blockIdx.x;
#pragma unroll
    for (int i = 0; i < 4; ++i) {
        int c = threadIdx.x + i * 256;
        out[(long)t * outStride + c] = (__bf16)in[(long)t * inStride + c];
    }
}

// out_bf16 = bf16(rmsnorm(S, g)) with output row stride
__global__ __launch_bounds__(256)
void rmsnorm_to_bf16_kernel(const float* __restrict__ S, const float* __restrict__ g,
                            __bf16* __restrict__ out, int outStride) {
    int t = blockIdx.x;
    float v[4], ss = 0.f;
#pragma unroll
    for (int i = 0; i < 4; ++i) {
        int c = threadIdx.x + i * 256;
        v[i] = S[(long)t * DMODEL + c];
        ss += v[i] * v[i];
    }
#pragma unroll
    for (int off = 16; off; off >>= 1) ss += __shfl_xor(ss, off, 32);
    __shared__ float red[8];
    if ((threadIdx.x & 31) == 0) red[threadIdx.x >> 5] = ss;
    __syncthreads();
    float tot = 0.f;
#pragma unroll
    for (int i = 0; i < 8; ++i) tot += red[i];
    float r = rsqrtf(tot * (1.f / DMODEL) + 1e-6f);
#pragma unroll
    for (int i = 0; i < 4; ++i) {
        int c = threadIdx.x + i * 256;
        out[(long)t * outStride + c] = (__bf16)(v[i] * r * g[c]);
    }
}

// S = rmsnorm(S + Y, g)  (f32, in place)
__global__ __launch_bounds__(256)
void add_rmsnorm_kernel(const float* __restrict__ Y, const float* __restrict__ g,
                        float* __restrict__ S) {
    int t = blockIdx.x;
    float v[4], ss = 0.f;
#pragma unroll
    for (int i = 0; i < 4; ++i) {
        int c = threadIdx.x + i * 256;
        v[i] = S[(long)t * DMODEL + c] + Y[(long)t * DMODEL + c];
        ss += v[i] * v[i];
    }
#pragma unroll
    for (int off = 16; off; off >>= 1) ss += __shfl_xor(ss, off, 32);
    __shared__ float red[8];
    if ((threadIdx.x & 31) == 0) red[threadIdx.x >> 5] = ss;
    __syncthreads();
    float tot = 0.f;
#pragma unroll
    for (int i = 0; i < 8; ++i) tot += red[i];
    float r = rsqrtf(tot * (1.f / DMODEL) + 1e-6f);
#pragma unroll
    for (int i = 0; i < 4; ++i) {
        int c = threadIdx.x + i * 256;
        S[(long)t * DMODEL + c] = v[i] * r * g[c];
    }
}

// Router: one wave per token. Writes logits (into d_out), topw/topi, atomics
// for load-balance means into acc[0..7]=sum probs, acc[8..15]=sum assign.
__global__ __launch_bounds__(256)
void router_kernel(const float* __restrict__ x, const float* __restrict__ rw,
                   float* __restrict__ logits, float* __restrict__ topw,
                   int* __restrict__ topi, float* __restrict__ acc) {
    int wave = threadIdx.x >> 5, lane = threadIdx.x & 31;
    int t = blockIdx.x * 8 + wave;
    float pa[NEXP] = {0.f, 0.f, 0.f, 0.f, 0.f, 0.f, 0.f, 0.f};
    for (int d = lane; d < DMODEL; d += 32) {
        float xv = x[(long)t * DMODEL + d];
#pragma unroll
        for (int e = 0; e < NEXP; ++e) pa[e] += xv * rw[e * DMODEL + d];
    }
#pragma unroll
    for (int e = 0; e < NEXP; ++e)
#pragma unroll
        for (int off = 16; off; off >>= 1) pa[e] += __shfl_xor(pa[e], off, 32);

    if (lane == 0) {
        float m = pa[0];
#pragma unroll
        for (int e = 1; e < NEXP; ++e) m = fmaxf(m, pa[e]);
        float p[NEXP], s = 0.f;
#pragma unroll
        for (int e = 0; e < NEXP; ++e) { p[e] = __expf(pa[e] - m); s += p[e]; }
        float inv = 1.f / s;
#pragma unroll
        for (int e = 0; e < NEXP; ++e) {
            p[e] *= inv;
            logits[(long)t * NEXP + e] = pa[e];
            atomicAdd(&acc[e], p[e]);
        }
        int i1 = 0;
#pragma unroll
        for (int e = 1; e < NEXP; ++e) if (p[e] > p[i1]) i1 = e;   // first max on ties
        int i2 = (i1 == 0) ? 1 : 0;
#pragma unroll
        for (int e = 0; e < NEXP; ++e) if (e != i1 && e != i2 && p[e] > p[i2]) i2 = e;
        float w0 = p[i1], w1 = p[i2], si = 1.f / (w0 + w1);
        topw[2 * t] = w0 * si; topw[2 * t + 1] = w1 * si;
        topi[2 * t] = i1;      topi[2 * t + 1] = i2;
        atomicAdd(&acc[8 + i1], 1.f);
        atomicAdd(&acc[8 + i2], 1.f);
    }
}

__global__ void lb_loss_kernel(const float* __restrict__ acc, float* __restrict__ out_lb) {
    if (threadIdx.x == 0 && blockIdx.x == 0) {
        float lb = 0.f;
        for (int e = 0; e < NEXP; ++e)
            lb += (acc[8 + e] * (1.f / T_TOK)) * (acc[e] * (1.f / T_TOK));
        *out_lb = (float)NEXP * lb;
    }
}

// out[t,:] += gate_weight(e) * S[t,:]
__global__ __launch_bounds__(256)
void combine_kernel(const float* __restrict__ S, const float* __restrict__ topw,
                    const int* __restrict__ topi, int e, float* __restrict__ out) {
    int t = blockIdx.x;
    float w = 0.f;
    if (topi[2 * t] == e)     w += topw[2 * t];
    if (topi[2 * t + 1] == e) w += topw[2 * t + 1];
    if (w != 0.f) {
#pragma unroll
        for (int i = 0; i < 4; ++i) {
            int c = threadIdx.x + i * 256;
            out[(long)t * DMODEL + c] += w * S[(long)t * DMODEL + c];
        }
    }
}

// ===========================================================================
// Host-side orchestration
// ===========================================================================
static void launch_gemm(int epi, const __bf16* A, const __bf16* W,
                        float* Cf, __bf16* Cb, int M, int N, int K, hipStream_t s) {
    dim3 grid(N / 128, M / 128), blk(256);
    if (epi == 0)      gemm_nt_wmma_kernel<0><<<grid, blk, 0, s>>>(A, W, Cf, Cb, M, N, K);
    else if (epi == 1) gemm_nt_wmma_kernel<1><<<grid, blk, 0, s>>>(A, W, Cf, Cb, M, N, K);
    else               gemm_nt_wmma_kernel<2><<<grid, blk, 0, s>>>(A, W, Cf, Cb, M, N, K);
}

extern "C" void kernel_launch(void* const* d_in, const int* in_sizes, int n_in,
                              void* d_out, int out_size, void* d_ws, size_t ws_size,
                              hipStream_t stream) {
    const float* x          = (const float*)d_in[0];   // [T, D]
    const float* router_w   = (const float*)d_in[1];   // [E, D]
    const float* adapter_w  = (const float*)d_in[2];   // [E, D, F]
    // d_in[3]=wq, d_in[4]=wk : do not affect output (len-1 attention), skipped.
    const float* wv         = (const float*)d_in[5];   // [E, D, D]
    const float* wo         = (const float*)d_in[6];   // [E, D, D]
    const float* g1         = (const float*)d_in[7];
    const float* g2         = (const float*)d_in[8];
    const float* g3         = (const float*)d_in[9];
    const float* g4         = (const float*)d_in[10];
    const float* mlp_w1     = (const float*)d_in[11];  // [E, F, D]
    const float* mlp_w2     = (const float*)d_in[12];  // [E, D, F]
    const float* init_state = (const float*)d_in[13];  // [E, T, D]

    float* out        = (float*)d_out;            // final [T*D] | lb_loss | logits [T*E]
    float* out_lb     = out + (long)T_TOK * DMODEL;
    float* out_logits = out_lb + 1;

    const long TD = (long)T_TOK * DMODEL;         // 4,194,304
    const long DF = (long)DMODEL * FDIM;          // 2,097,152 (per expert)
    const long DD = (long)DMODEL * DMODEL;        // 1,048,576 (per expert)

    // ---- workspace carve-out (256B aligned) ----
    char* p = (char*)d_ws;
    auto alloc = [&](size_t bytes) { void* r = (void*)p; p += (bytes + 255) & ~(size_t)255; return r; };
    __bf16* adapter_b = (__bf16*)alloc((size_t)NEXP * DF * 2);
    __bf16* wv_b      = (__bf16*)alloc((size_t)NEXP * DD * 2);
    __bf16* wo_b      = (__bf16*)alloc((size_t)NEXP * DD * 2);
    __bf16* w1_b      = (__bf16*)alloc((size_t)NEXP * DF * 2);
    __bf16* w2_b      = (__bf16*)alloc((size_t)NEXP * DF * 2);
    __bf16* comb_b    = (__bf16*)alloc((size_t)T_TOK * FDIM * 2);   // [state | x]
    __bf16* s_b       = (__bf16*)alloc((size_t)TD * 2);
    __bf16* H_b       = (__bf16*)alloc((size_t)T_TOK * FDIM * 2);
    __bf16* v_b       = (__bf16*)alloc((size_t)TD * 2);
    float*  S_f       = (float*)alloc((size_t)TD * 4);
    float*  Y_f       = (float*)alloc((size_t)TD * 4);
    float*  topw      = (float*)alloc((size_t)T_TOK * 2 * 4);
    int*    topi      = (int*)alloc((size_t)T_TOK * 2 * 4);
    float*  accp      = (float*)alloc(64);

    // ---- init: zero combine target + lb accumulators ----
    zero_f32_kernel<<<2048, 256, 0, stream>>>(out, TD);
    zero_f32_kernel<<<1, 32, 0, stream>>>(accp, 16);

    // ---- router + load-balance loss ----
    router_kernel<<<T_TOK / 8, 256, 0, stream>>>(x, router_w, out_logits, topw, topi, accp);
    lb_loss_kernel<<<1, 32, 0, stream>>>(accp, out_lb);

    // ---- one-time f32 -> bf16 weight conversion ----
    f32_to_bf16_kernel<<<2048, 256, 0, stream>>>(adapter_w, adapter_b, (long)NEXP * DF);
    f32_to_bf16_kernel<<<2048, 256, 0, stream>>>(wv, wv_b, (long)NEXP * DD);
    f32_to_bf16_kernel<<<2048, 256, 0, stream>>>(wo, wo_b, (long)NEXP * DD);
    f32_to_bf16_kernel<<<2048, 256, 0, stream>>>(mlp_w1, w1_b, (long)NEXP * DF);
    f32_to_bf16_kernel<<<2048, 256, 0, stream>>>(mlp_w2, w2_b, (long)NEXP * DF);

    // static half of `combined` (same for every expert/step): comb[:,1024:] = bf16(x)
    conv_rows_kernel<<<T_TOK, 256, 0, stream>>>(x, DMODEL, comb_b + DMODEL, FDIM);

    // ---- densified experts ----
    for (int e = 0; e < NEXP; ++e) {
        copy_f32_kernel<<<2048, 256, 0, stream>>>(init_state + (long)e * TD, S_f, TD);
        for (int step = 0; step < 2; ++step) {
            // combined[:, :1024] = bf16(state)
            conv_rows_kernel<<<T_TOK, 256, 0, stream>>>(S_f, DMODEL, comb_b, FDIM);
            // state = combined @ adapter^T  (res kept in S_f)
            launch_gemm(0, comb_b, adapter_b + (long)e * DF, S_f, nullptr,
                        T_TOK, DMODEL, FDIM, stream);
            // s = rmsnorm(state, g1)
            rmsnorm_to_bf16_kernel<<<T_TOK, 256, 0, stream>>>(S_f, g1 + e * DMODEL, s_b, DMODEL);
            // attn = (s @ wv^T) @ wo^T   (len-1 softmax == 1)
            launch_gemm(1, s_b, wv_b + (long)e * DD, nullptr, v_b, T_TOK, DMODEL, DMODEL, stream);
            launch_gemm(0, v_b, wo_b + (long)e * DD, Y_f, nullptr, T_TOK, DMODEL, DMODEL, stream);
            // state = rmsnorm(res + attn, g2)
            add_rmsnorm_kernel<<<T_TOK, 256, 0, stream>>>(Y_f, g2 + e * DMODEL, S_f);
            // s = rmsnorm(state, g3)
            rmsnorm_to_bf16_kernel<<<T_TOK, 256, 0, stream>>>(S_f, g3 + e * DMODEL, s_b, DMODEL);
            // h = silu(s @ w1^T); m = h @ w2^T
            launch_gemm(2, s_b, w1_b + (long)e * DF, nullptr, H_b, T_TOK, FDIM, DMODEL, stream);
            launch_gemm(0, H_b, w2_b + (long)e * DF, Y_f, nullptr, T_TOK, DMODEL, FDIM, stream);
            // state = rmsnorm(res + m, g4)
            add_rmsnorm_kernel<<<T_TOK, 256, 0, stream>>>(Y_f, g4 + e * DMODEL, S_f);
        }
        // gather/scale into final output
        combine_kernel<<<T_TOK, 256, 0, stream>>>(S_f, topw, topi, e, out);
    }
}
